// PsiCMPSCell_59287728554685
// MI455X (gfx1250) — matrix operand order repaired and checked by reading, not compiled, not run on hardware
//
#include <hip/hip_runtime.h>

// ---------------- CDNA5 types ----------------
typedef __attribute__((ext_vector_type(16))) _Float16 v16h;
typedef __attribute__((ext_vector_type(8)))  float    v8f;
typedef __attribute__((ext_vector_type(2)))  __fp16   v2hf; // matches cvt_pkrtz return type

#define D_    64
#define NB_   256
#define T_    4096
#define DT_   0.001f
#define AA_   100.0f
#define ALPHA_ (-0.0005f)            /* -DT*SIGMA^2/2 */
#define PRE_   (-20.0f)              /* 1/(A*ALPHA) */
#define EPS_   1e-12f

union FragU { v16h v; unsigned u[8]; };
union H2U   { v2hf h2; unsigned u; };

static __device__ __forceinline__ unsigned pkrtz(float a, float b) {
  H2U x; x.h2 = __builtin_amdgcn_cvt_pkrtz(a, b); return x.u;
}
static __device__ __forceinline__ unsigned swapx16u(unsigned x) {
  return (unsigned)__builtin_amdgcn_ds_swizzle((int)x, 0x401F); // xor 0x10, and 0x1f
}
static __device__ __forceinline__ float swapx16f(float x) {
  return __int_as_float(__builtin_amdgcn_ds_swizzle(__float_as_int(x), 0x401F));
}
static __device__ __forceinline__ v8f wmma16(v16h a, v16h b, v8f c) {
  return __builtin_amdgcn_wmma_f32_16x16x32_f16(false, a, false, b, (short)0, c, false, false);
}

// Pack CT-layout f32 (lane<->batch, VGPR v<->d = mt*16+8h+v) into f16 K-pairs.
static __device__ __forceinline__ void pack_ct(const float c[4][8], unsigned pk[4][4]) {
#pragma unroll
  for (int mt = 0; mt < 4; ++mt)
#pragma unroll
    for (int j = 0; j < 4; ++j)
      pk[mt][j] = pkrtz(c[mt][2 * j], c[mt][2 * j + 1]);
}

// Build B-fragments (16x16x32, K=d) for both k-chunks from packed pairs.
// B lane<16 holds K=kc*32+[0..15]; lane>=16 holds K=kc*32+[16..31].
static __device__ __forceinline__ void frags_from_pk(const unsigned pk[4][4], bool up, v16h Bf[2]) {
#pragma unroll
  for (int kc = 0; kc < 2; ++kc) {
    FragU f;
#pragma unroll
    for (int j = 0; j < 4; ++j) {
      unsigned lo = pk[2 * kc][j];      // h=0 pairs (own for lanes<16)
      unsigned hi = pk[2 * kc + 1][j];  // h=1 pairs (own for lanes>=16)
      unsigned s1 = swapx16u(hi);       // lanes>=16 <- partner's pk[2kc+1][j]
      unsigned s2 = swapx16u(lo);       // lanes<16  <- partner's pk[2kc][j]
      f.u[j]     = up ? s1 : lo;
      f.u[j + 4] = up ? hi : s2;
    }
    Bf[kc] = f.v;
  }
}

static __device__ __forceinline__ void build_bfrags(const float cr[4][8], const float ci[4][8],
                                                    bool up, v16h Br[2], v16h Bi[2]) {
  unsigned pkr[4][4], pki[4][4];
  pack_ct(cr, pkr);
  pack_ct(ci, pki);
  frags_from_pk(pkr, up, Br);
  frags_from_pk(pki, up, Bi);
}

// Complex C += A(orientation O, from LDS f16 frags) * B.  Output in CT layout.
template <int O>
static __device__ __forceinline__ void cplx_mm(const unsigned* __restrict__ af, int lane,
                                               const v16h Br[2], const v16h Bi[2],
                                               float Cr[4][8], float Ci[4][8]) {
  v16h Bn[2]; // -Bi  (f16 WMMA NEG on A/B must be 0 per ISA; negate via sign xor)
#pragma unroll
  for (int kc = 0; kc < 2; ++kc) {
    FragU x; x.v = Bi[kc];
#pragma unroll
    for (int j = 0; j < 8; ++j) x.u[j] ^= 0x80008000u;
    Bn[kc] = x.v;
  }
#pragma unroll
  for (int mt = 0; mt < 4; ++mt) {
    const unsigned* base = af + ((O * 4 + mt) * 4) * 256 + lane * 8;
    v16h Ar0 = *(const v16h*)(base);
    v16h Ai0 = *(const v16h*)(base + 256);
    v16h Ar1 = *(const v16h*)(base + 512);
    v16h Ai1 = *(const v16h*)(base + 768);
    v8f cr, ci;
#pragma unroll
    for (int v = 0; v < 8; ++v) { cr[v] = Cr[mt][v]; ci[v] = Ci[mt][v]; }
    cr = wmma16(Ar0, Br[0], cr);
    cr = wmma16(Ai0, Bn[0], cr);
    cr = wmma16(Ar1, Br[1], cr);
    cr = wmma16(Ai1, Bn[1], cr);
    ci = wmma16(Ar0, Bi[0], ci);
    ci = wmma16(Ai0, Br[0], ci);
    ci = wmma16(Ar1, Bi[1], ci);
    ci = wmma16(Ai1, Br[1], ci);
#pragma unroll
    for (int v = 0; v < 8; ++v) { Cr[mt][v] = cr[v]; Ci[mt][v] = ci[v]; }
  }
}

__global__ __launch_bounds__(32) void cmps_scan_kernel(
    const float* __restrict__ sig, const float* __restrict__ Rx, const float* __restrict__ Ry,
    const float* __restrict__ freqs, const float* __restrict__ px, const float* __restrict__ py,
    float* __restrict__ out) {
  // A-fragments: [orient(2)][mt(4)][kc(2)][comp(2)] x 256 u32 (per-lane 32B blobs) = 32KB
  __shared__ alignas(32) unsigned afrag[32 * 256];
  __shared__ alignas(16) float elds[32 * 64]; // per-lane E twiddle blob: 32 er then 32 ei
  __shared__ float stage[4 * 64];             // dRx | dRy | psi_x | psi_y

  const int lane = threadIdx.x;
  const int m = lane & 15;
  const int h = lane >> 4;
  const bool up = (h == 1);
  const int wg = blockIdx.x;

  // ---- setup: diagonals + psi staging ----
  stage[lane]           = Rx[lane * 65];
  stage[lane + 32]      = Rx[(lane + 32) * 65];
  stage[64 + lane]      = Ry[lane * 65];
  stage[64 + lane + 32] = Ry[(lane + 32) * 65];
  stage[128 + lane]      = px[lane];
  stage[128 + lane + 32] = px[lane + 32];
  stage[192 + lane]      = py[lane];
  stage[192 + lane + 32] = py[lane + 32];
  __syncthreads();

  // ---- A fragments: o=0 -> R' (M=b,K=c); o=1 -> conj(R')^T (M=b,K=c) ----
#pragma unroll 1
  for (int o = 0; o < 2; ++o)
#pragma unroll 1
    for (int mt = 0; mt < 4; ++mt)
      for (int kc = 0; kc < 2; ++kc)
        for (int comp = 0; comp < 2; ++comp) {
          const int blk = ((o * 4 + mt) * 2 + kc) * 2 + comp;
          const float* M  = comp ? Ry : Rx;
          const float* dg = comp ? (stage + 64) : stage;
          for (int j = 0; j < 8; ++j) {
            int K = 16 * h + (j << 1); // pair base within chunk
            int row = mt * 16 + m;
            int c0 = kc * 32 + K;
            float v0, v1;
            if (o == 0) { // R'[row][c] = M[row][c] - M[c][c]
              v0 = M[row * 64 + c0]     - dg[c0];
              v1 = M[row * 64 + c0 + 1] - dg[c0 + 1];
            } else {      // conj(R')^T[row][c] = +/-(M[c][row] - M[row][row])
              v0 = M[c0 * 64 + row]       - dg[row];
              v1 = M[(c0 + 1) * 64 + row] - dg[row];
              if (comp) { v0 = -v0; v1 = -v1; }
            }
            afrag[blk * 256 + lane * 8 + j] = pkrtz(v0, v1);
          }
        }

  // ---- E twiddle blob (constant per-step rotation exp(-i f DT)) ----
  for (int mt = 0; mt < 4; ++mt)
    for (int v = 0; v < 8; ++v) {
      int d = mt * 16 + 8 * h + v;
      float a = freqs[d] * DT_;
      elds[lane * 64 + mt * 8 + v]      = cosf(a);
      elds[lane * 64 + 32 + mt * 8 + v] = -sinf(a);
    }

  // ---- initial w = normalize(psi_x + i psi_y) broadcast over batch (CT layout) ----
  float q0 = 0.f;
  for (int k = 0; k < 64; ++k) {
    float a = stage[128 + k], b = stage[192 + k];
    q0 += a * a + b * b;
  }
  float inv0 = __builtin_amdgcn_rsqf(fmaxf(q0, EPS_));
  float wr[4][8], wi[4][8];
#pragma unroll
  for (int mt = 0; mt < 4; ++mt)
#pragma unroll
    for (int v = 0; v < 8; ++v) {
      int d = mt * 16 + 8 * h + v;
      wr[mt][v] = stage[128 + d] * inv0;
      wi[mt][v] = stage[192 + d] * inv0;
    }

  v16h Bwr[2], Bwi[2];
  build_bfrags(wr, wi, up, Bwr, Bwi);

  float s_cur = sig[wg * 16 + m];
  float acc = 0.f;

#pragma unroll 1
  for (int t = 0; t < T_; ++t) {
    int tn = (t + 1 < T_) ? (t + 1) : t;
    float s_next = sig[tn * NB_ + wg * 16 + m]; // prefetch next signal

    // --- matmul1: RU^T = R' * w^T ---
    float Cr[4][8], Ci[4][8];
#pragma unroll
    for (int mt = 0; mt < 4; ++mt)
#pragma unroll
      for (int v = 0; v < 8; ++v) { Cr[mt][v] = 0.f; Ci[mt][v] = 0.f; }
    cplx_mm<0>(afrag, lane, Bwr, Bwi, Cr, Ci);

    // B-fragments of RU^T for matmul2
    v16h BRr[2], BRi[2];
    build_bfrags(Cr, Ci, up, BRr, BRi);

    // --- C2 = (s/(A*alpha))*RU ;  C2 += conj(R')^T * RU^T  => dU = alpha*C2 ---
    float pre = s_cur * PRE_;
    float C2r[4][8], C2i[4][8];
#pragma unroll
    for (int mt = 0; mt < 4; ++mt)
#pragma unroll
      for (int v = 0; v < 8; ++v) { C2r[mt][v] = pre * Cr[mt][v]; C2i[mt][v] = pre * Ci[mt][v]; }
    cplx_mm<1>(afrag, lane, BRr, BRi, C2r, C2i);

    // --- w2 = w + alpha*C2 (unnormalized, pre-loss state) ---
#pragma unroll
    for (int mt = 0; mt < 4; ++mt)
#pragma unroll
      for (int v = 0; v < 8; ++v) {
        wr[mt][v] = fmaf(ALPHA_, C2r[mt][v], wr[mt][v]);
        wi[mt][v] = fmaf(ALPHA_, C2i[mt][v], wi[mt][v]);
      }

    // --- matmul3: V^T = R' * w2^T ---
    v16h B2r[2], B2i[2];
    build_bfrags(wr, wi, up, B2r, B2i);
    float Vr[4][8], Vi[4][8];
#pragma unroll
    for (int mt = 0; mt < 4; ++mt)
#pragma unroll
      for (int v = 0; v < 8; ++v) { Vr[mt][v] = 0.f; Vi[mt][v] = 0.f; }
    cplx_mm<0>(afrag, lane, B2r, B2i, Vr, Vi);

    // --- loss: exp = 2*Re(conj(w2) . V) ; loss -= log(1 + exp*s/A) ---
    float p = 0.f;
#pragma unroll
    for (int mt = 0; mt < 4; ++mt)
#pragma unroll
      for (int v = 0; v < 8; ++v)
        p = fmaf(wr[mt][v], Vr[mt][v], fmaf(wi[mt][v], Vi[mt][v], p));
    p += swapx16f(p);
    float expv = 2.f * p;
    acc += __logf(1.f + expv * (s_cur * (1.f / AA_)));

    // --- normalize rows and apply constant twiddle E ---
    float q = 0.f;
#pragma unroll
    for (int mt = 0; mt < 4; ++mt)
#pragma unroll
      for (int v = 0; v < 8; ++v)
        q = fmaf(wr[mt][v], wr[mt][v], fmaf(wi[mt][v], wi[mt][v], q));
    q += swapx16f(q);
    float inv = __builtin_amdgcn_rsqf(fmaxf(q, EPS_));

#pragma unroll
    for (int mt = 0; mt < 4; ++mt)
#pragma unroll
      for (int v = 0; v < 8; ++v) {
        float er = elds[lane * 64 + mt * 8 + v];
        float ei = elds[lane * 64 + 32 + mt * 8 + v];
        float a = wr[mt][v], b = wi[mt][v];
        wr[mt][v] = inv * (a * er - b * ei);
        wi[mt][v] = inv * (a * ei + b * er);
      }

    // fragments of w for next step's matmul1
    build_bfrags(wr, wi, up, Bwr, Bwi);
    s_cur = s_next;
  }

  if (lane < 16) out[wg * 16 + lane] = -acc;
}

extern "C" void kernel_launch(void* const* d_in, const int* in_sizes, int n_in,
                              void* d_out, int out_size, void* d_ws, size_t ws_size,
                              hipStream_t stream) {
  const float* sig = (const float*)d_in[0];
  const float* Rx  = (const float*)d_in[1];
  const float* Ry  = (const float*)d_in[2];
  const float* fr  = (const float*)d_in[3];
  const float* ppx = (const float*)d_in[4];
  const float* ppy = (const float*)d_in[5];
  float* out = (float*)d_out;
  hipLaunchKernelGGL(cmps_scan_kernel, dim3(NB_ / 16), dim3(32), 0, stream,
                     sig, Rx, Ry, fr, ppx, ppy, out);
}
